// MLPSimilarityAttention_69303592288275
// MI455X (gfx1250) — matrix operand order repaired and checked by reading, not compiled, not run on hardware
//
#include <hip/hip_runtime.h>

#define D_  256
#define H1_ 256
#define H2_ 128
#define B_  2
#define LQ  512
#define LK  512

typedef __attribute__((ext_vector_type(16))) _Float16 v16h;
typedef __attribute__((ext_vector_type(8)))  float    v8f;

// ---------------------------------------------------------------------------
// Kernel 1: projections.  blockIdx.x: 0 = query path (adds b1), 1 = key path.
// ---------------------------------------------------------------------------
__global__ void proj_kernel(const float* __restrict__ query,
                            const float* __restrict__ key,
                            const float* __restrict__ W1,
                            const float* __restrict__ b1,
                            float* __restrict__ qph,
                            float* __restrict__ kp)
{
    __shared__ float row[D_];
    const int isKey = blockIdx.x;
    const int r     = blockIdx.y;
    const int b     = blockIdx.z;
    const int h     = threadIdx.x;            // 256 threads == H1
    const float* src = isKey ? key : query;
    const float* w   = W1 + (isKey ? (size_t)D_ * H1_ : 0);
    row[h] = src[((size_t)b * LQ + r) * D_ + h];
    __syncthreads();
    float acc = isKey ? 0.0f : b1[h];
#pragma unroll 8
    for (int d = 0; d < D_; ++d)
        acc += row[d] * w[(size_t)d * H1_ + h];
    float* dst = isKey ? kp : qph;
    dst[((size_t)b * LQ + r) * H1_ + h] = acc;
}

// ---------------------------------------------------------------------------
// Kernel 2: W2t[n][k] = (f16) W2[k][n]   (128 x 256 f16, row-major in n)
// ---------------------------------------------------------------------------
__global__ void w2t_kernel(const float* __restrict__ W2, _Float16* __restrict__ W2t)
{
    int i = blockIdx.x * blockDim.x + threadIdx.x;    // [0, 32768)
    int n = i >> 8;
    int k = i & 255;
    W2t[i] = (_Float16)W2[(size_t)k * H2_ + n];
}

// ---------------------------------------------------------------------------
// Build 8 A-fragments (h1 = relu(qp+b1+kp) in f16, exact 16x32 WMMA layout)
// ---------------------------------------------------------------------------
__device__ __forceinline__ void build_A(v16h* A, const float* __restrict__ kprow,
                                        const float* qs, int dlt)
{
#pragma unroll
    for (int kt = 0; kt < 8; ++kt) {
        const int bh = kt * 32 + dlt;
        const float4 lo0 = *(const float4*)(kprow + bh);
        const float4 lo1 = *(const float4*)(kprow + bh + 4);
        const float4 hi0 = *(const float4*)(kprow + bh + 16);
        const float4 hi1 = *(const float4*)(kprow + bh + 20);
        float v[16];
        v[0]=lo0.x;  v[1]=lo0.y;  v[2]=lo0.z;  v[3]=lo0.w;
        v[4]=lo1.x;  v[5]=lo1.y;  v[6]=lo1.z;  v[7]=lo1.w;
        v[8]=hi0.x;  v[9]=hi0.y;  v[10]=hi0.z; v[11]=hi0.w;
        v[12]=hi1.x; v[13]=hi1.y; v[14]=hi1.z; v[15]=hi1.w;
#pragma unroll
        for (int j = 0; j < 8; ++j)
            A[kt][j] = (_Float16)fmaxf(v[j] + qs[bh + j], 0.0f);
#pragma unroll
        for (int j = 0; j < 8; ++j)
            A[kt][8 + j] = (_Float16)fmaxf(v[8 + j] + qs[bh + 16 + j], 0.0f);
    }
}

// ---------------------------------------------------------------------------
// Kernel 3: fused h1 -> h2 -> score with WMMA.
// grid = (LK/256, LQ, B), block = 256 (8 waves); wave handles 32 k's (2 M-tiles).
// W2t staged in LDS (async copy), padded rows: 33 uint4 (528 B) per 256-col row
// -> dword stride 132 = 4 (mod 64): bank-conflict-free ds_load_b128 fragments.
// ---------------------------------------------------------------------------
__global__ void __launch_bounds__(256)
scores_kernel(const float* __restrict__ qph,
              const float* __restrict__ kp,
              const _Float16* __restrict__ W2t,
              const float* __restrict__ b2,
              const float* __restrict__ W3,
              const float* __restrict__ b3,
              float* __restrict__ scores)
{
    __shared__ uint4 Wl[H2_ * 33];   // 67,584 B padded W2t (f16)
    __shared__ float qs[H1_];        // qp + b1 row

    const int b    = blockIdx.z;
    const int q    = blockIdx.y;
    const int tid  = threadIdx.x;
    const int wave = tid >> 5;
    const int lane = tid & 31;

    // ---- async-stage W2t (64 KB) into padded LDS: gfx1250 ASYNC engine ----
    {
#pragma unroll
        for (int i = 0; i < 16; ++i) {
            const int id = tid + i * 256;            // uint4 chunk id [0,4096)
            const int r  = id >> 5;                  // row   [0,128)
            const int c  = id & 31;                  // chunk [0,32)
            unsigned ldsOff = (unsigned)(uintptr_t)(&Wl[r * 33 + c]);
            unsigned gOff   = (unsigned)(id * 16);   // byte offset in W2t
            asm volatile("global_load_async_to_lds_b128 %0, %1, %2"
                         :: "v"(ldsOff), "v"(gOff), "s"(W2t) : "memory");
        }
        asm volatile("s_wait_asynccnt 0" ::: "memory");
    }
    qs[tid] = qph[((size_t)b * LQ + q) * H1_ + tid];
    __syncthreads();

    const int m    = lane & 15;      // row within a 16-k tile
    const int half = lane >> 4;      // 0: K {0..7,16..23}; 1: K {8..15,24..31}
    const int dlt  = half * 8;
    const int k0   = blockIdx.x * 256 + wave * 32;
    const float* kprow0 = kp + ((size_t)b * LK + (k0 + m)) * H1_;
    const float* kprow1 = kprow0 + (size_t)16 * H1_;

    v16h A0[8], A1[8];
    build_A(A0, kprow0, qs, dlt);
    build_A(A1, kprow1, qs, dlt);

    float s0[8], s1[8];
#pragma unroll
    for (int r = 0; r < 8; ++r) { s0[r] = 0.0f; s1[r] = 0.0f; }

    const int n = lane & 15;
#pragma unroll
    for (int nt = 0; nt < 8; ++nt) {
        const int ng = nt * 16 + n;
        v8f acc0 = {}, acc1 = {};
#pragma unroll
        for (int kt = 0; kt < 8; ++kt) {
            union { uint4 u[2]; v16h h; } bu;
            const int ci = ng * 33 + kt * 4 + half;
            bu.u[0] = Wl[ci];
            bu.u[1] = Wl[ci + 2];
            acc0 = __builtin_amdgcn_wmma_f32_16x16x32_f16(
                       false, A0[kt], false, bu.h, (short)0, acc0, false, false);
            acc1 = __builtin_amdgcn_wmma_f32_16x16x32_f16(
                       false, A1[kt], false, bu.h, (short)0, acc1, false, false);
        }
        const float bb = b2[ng];
        const float w3 = W3[ng];
#pragma unroll
        for (int r = 0; r < 8; ++r) {
            s0[r] += fmaxf(acc0[r] + bb, 0.0f) * w3;
            s1[r] += fmaxf(acc1[r] + bb, 0.0f) * w3;
        }
    }

    // ---- reduce across the 16-lane half (sum over N) and store scores ----
    const float b3v = b3[0];
#pragma unroll
    for (int r = 0; r < 8; ++r) {
        float v0 = s0[r], v1 = s1[r];
        v0 += __shfl_xor(v0, 1, 32);  v1 += __shfl_xor(v1, 1, 32);
        v0 += __shfl_xor(v0, 2, 32);  v1 += __shfl_xor(v1, 2, 32);
        v0 += __shfl_xor(v0, 4, 32);  v1 += __shfl_xor(v1, 4, 32);
        v0 += __shfl_xor(v0, 8, 32);  v1 += __shfl_xor(v1, 8, 32);
        s0[r] = v0; s1[r] = v1;
    }
    if ((lane & 15) == 0) {
        float* srow = scores + ((size_t)b * LQ + q) * LK;
#pragma unroll
        for (int r = 0; r < 8; ++r) {
            srow[k0 + r + half * 8]      = s0[r] + b3v;
            srow[k0 + 16 + r + half * 8] = s1[r] + b3v;
        }
    }
}

// ---------------------------------------------------------------------------
// Kernel 4: masked softmax over k, in place (scores buffer == attn output).
// ---------------------------------------------------------------------------
__global__ void softmax_kernel(float* __restrict__ attn,
                               const int* __restrict__ mask)
{
    __shared__ float red[256];
    const int row = blockIdx.x;               // b*LQ + q
    const int tid = threadIdx.x;              // 256 threads, 2 elems each
    float* srow = attn + (size_t)row * LK;
    const int* mrow = mask + (size_t)row * LK;
    float x0 = srow[tid], x1 = srow[tid + 256];
    if (mrow[tid]       == 0) x0 = -1.0e9f;
    if (mrow[tid + 256] == 0) x1 = -1.0e9f;
    red[tid] = fmaxf(x0, x1);
    __syncthreads();
    for (int s = 128; s > 0; s >>= 1) {
        if (tid < s) red[tid] = fmaxf(red[tid], red[tid + s]);
        __syncthreads();
    }
    const float mx = red[0];
    __syncthreads();
    float e0 = __expf(x0 - mx), e1 = __expf(x1 - mx);
    red[tid] = e0 + e1;
    __syncthreads();
    for (int s = 128; s > 0; s >>= 1) {
        if (tid < s) red[tid] += red[tid + s];
        __syncthreads();
    }
    const float inv = 1.0f / red[0];
    srow[tid]       = e0 * inv;
    srow[tid + 256] = e1 * inv;
}

// ---------------------------------------------------------------------------
// Kernel 5: out[b][q][d] = sum_k attn[b][q][k] * value[b][k][d]
// ---------------------------------------------------------------------------
__global__ void av_kernel(const float* __restrict__ attn,
                          const float* __restrict__ value,
                          float* __restrict__ out)
{
    __shared__ float arow[LK];
    const int row = blockIdx.x;               // b*LQ + q
    const int b   = row / LQ;
    const int d   = threadIdx.x;              // 256 == D
    arow[d]       = attn[(size_t)row * LK + d];
    arow[d + 256] = attn[(size_t)row * LK + d + 256];
    __syncthreads();
    const float* vb = value + (size_t)b * LK * D_;
    float acc = 0.0f;
#pragma unroll 4
    for (int k = 0; k < LK; ++k)
        acc += arow[k] * vb[(size_t)k * D_ + d];
    out[(size_t)row * D_ + d] = acc;
}

// ---------------------------------------------------------------------------
extern "C" void kernel_launch(void* const* d_in, const int* in_sizes, int n_in,
                              void* d_out, int out_size, void* d_ws, size_t ws_size,
                              hipStream_t stream)
{
    const float* query = (const float*)d_in[0];
    const float* key_t = (const float*)d_in[1];
    const float* value = (const float*)d_in[2];
    const int*   mask  = (const int*)  d_in[3];
    const float* W1    = (const float*)d_in[4];
    const float* b1    = (const float*)d_in[5];
    const float* W2    = (const float*)d_in[6];
    const float* b2    = (const float*)d_in[7];
    const float* W3    = (const float*)d_in[8];
    const float* b3    = (const float*)d_in[9];

    // d_out = [ out (B*LQ*D) | attn (B*LQ*LK) ]
    float* out  = (float*)d_out;
    float* attn = out + (size_t)B_ * LQ * D_;     // scores computed in place here

    // workspace: qph (1MB) | kp (1MB) | W2t f16 (64KB)
    float*    qph = (float*)d_ws;
    float*    kp  = qph + (size_t)B_ * LQ * H1_;
    _Float16* W2t = (_Float16*)(kp + (size_t)B_ * LK * H1_);

    proj_kernel   <<<dim3(2, LQ, B_),        256, 0, stream>>>(query, key_t, W1, b1, qph, kp);
    w2t_kernel    <<<dim3(H1_ * H2_ / 256),  256, 0, stream>>>(W2, W2t);
    scores_kernel <<<dim3(LK / 256, LQ, B_), 256, 0, stream>>>(qph, kp, W2t, b2, W3, b3, attn);
    softmax_kernel<<<dim3(B_ * LQ),          256, 0, stream>>>(attn, mask);
    av_kernel     <<<dim3(B_ * LQ),          256, 0, stream>>>(attn, value, out);
}
